// LocalWindowAttention_403726926577
// MI455X (gfx1250) — matrix operand order repaired and checked by reading, not compiled, not run on hardware
//
#include <hip/hip_runtime.h>

// ---------------------------------------------------------------------------
// Fused local-window attention for MI455X (gfx1250, wave32, WMMA bf16).
// One workgroup (8 wave32s) per 8x8 window. All three matmul stages run on
// v_wmma_f32_16x16x32_bf16 with fp32 accumulation; softmax in fp32 VALU.
// v2: register double-buffered k-loop (overlap loads with WMMA), A-fragment
//     reuse across 4-6 accumulators per wave, 256-thread shuffle softmax,
//     biases staged in LDS.
// ---------------------------------------------------------------------------

typedef __attribute__((ext_vector_type(16))) __bf16 v16bf;
typedef __attribute__((ext_vector_type(8)))  __bf16 v8bf;
typedef __attribute__((ext_vector_type(8)))  float  v8f;

#define B_IMG   8
#define C_DIM   192
#define H_IMG   256
#define W_IMG   256
#define WS      8
#define NH      4
#define HD      48
#define TT      64            // tokens per window = WS*WS
#define QKV_ROWS (3 * C_DIM)  // 576
#define SCALE   0.14433756729740643f  // 48^-0.5

// ----- WMMA fragment loaders (ISA 7.12.2 layouts) ---------------------------
// A matrix 16x32 bf16, row-major source with leading dim `ld`:
//   lane = 16*hh + m ; regs 0..7  -> A[m][k0 + 8*hh + 0..7]
//                      regs 8..15 -> A[m][k0 + 16 + 8*hh + 0..7]
__device__ __forceinline__ v16bf load_a_frag(const __bf16* base, int row0,
                                             int k0, int ld, int lane) {
  const int m  = lane & 15;
  const int hh = lane >> 4;
  const __bf16* p = base + (size_t)(row0 + m) * ld + k0 + hh * 8;
  union { v16bf v; v8bf h[2]; } u;
  u.h[0] = *(const v8bf*)(p);
  u.h[1] = *(const v8bf*)(p + 16);
  return u.v;
}

// B matrix 32x16 bf16, sourced from a TRANSPOSED ([N][K]) buffer with leading
// dim `ld`: lane = 16*hh + n ; regs 0..7  -> B[k0 + 8*hh + 0..7][n]
//                              regs 8..15 -> B[k0 + 16 + 8*hh + 0..7][n]
__device__ __forceinline__ v16bf load_b_frag(const __bf16* baseT, int col0,
                                             int k0, int ld, int lane) {
  const int n  = lane & 15;
  const int hh = lane >> 4;
  const __bf16* p = baseT + (size_t)(col0 + n) * ld + k0 + hh * 8;
  union { v16bf v; v8bf h[2]; } u;
  u.h[0] = *(const v8bf*)(p);
  u.h[1] = *(const v8bf*)(p + 16);
  return u.v;
}

#define WMMA_BF16(A, B, Cacc)                                                  \
  __builtin_amdgcn_wmma_f32_16x16x32_bf16(false, (A), false, (B), (short)0,   \
                                          (Cacc), false, false)

// ----- dynamic LDS layout (~150 KB / workgroup, < 320 KB WGP budget) --------
struct __align__(16) Smem {
  union {
    __bf16 xT[TT * C_DIM];     // window tile transposed [t][c] (B^T for QKV)
    float  s[TT * 65];         // attention scores, padded stride (per head)
  } u;
  __bf16 qT[NH][TT * 64];      // Q^T per head [t][d], d padded 48->64, scaled
  __bf16 kT[NH][TT * 64];      // K^T per head [t][d] (B^T for Q^T K)
  __bf16 vv[NH][HD * TT];      // V per head [d][t]   (B^T for P V^T)
  __bf16 p [TT * TT];          // softmax probs [tq][tk]
  __bf16 attn[TT * C_DIM];     // attention output [t][c] (A of final GEMM)
  float  bq[QKV_ROWS];         // qkv bias staged in LDS
  float  bo[C_DIM];            // out bias staged in LDS
};

// ----- kernel 0: fp32 -> bf16 weight conversion into workspace --------------
__global__ void convert_weights_kernel(const float* __restrict__ wq,
                                       const float* __restrict__ wo,
                                       __bf16* __restrict__ wq_b,
                                       __bf16* __restrict__ wo_b) {
  int i = blockIdx.x * 256 + threadIdx.x;
  if (i < QKV_ROWS * C_DIM) wq_b[i] = (__bf16)wq[i];
  if (i < C_DIM * C_DIM)    wo_b[i] = (__bf16)wo[i];
}

// ----- kernel 1: fused window attention -------------------------------------
__global__ void __launch_bounds__(256)
win_attn_kernel(const float* __restrict__ x,
                const float* __restrict__ b_qkv,
                const float* __restrict__ b_out,
                const __bf16* __restrict__ wq,   // [576][192] bf16
                const __bf16* __restrict__ wo,   // [192][192] bf16 (== B^T)
                float* __restrict__ out) {
  extern __shared__ __align__(16) char smem_raw[];
  Smem& sm = *reinterpret_cast<Smem*>(smem_raw);

  const int tid  = threadIdx.x;
  const int lane = tid & 31;
  const int wave = tid >> 5;
  const int fn   = lane & 15;   // fragment col (n) / row (m) lane id
  const int fh   = lane >> 4;   // fragment half

  const int blk  = blockIdx.x;          // 8192 windows
  const int bimg = blk >> 10;           // / (32*32)
  const int rw   = blk & 1023;
  const int h0   = (rw >> 5) * WS;
  const int w0   = (rw & 31) * WS;

  // ---- Phase 1: stage window tile x^T[t][c] + biases into LDS ----
  for (int i = tid; i < C_DIM * TT; i += 256) {
    const int c = i >> 6;               // consecutive lanes -> consecutive t
    const int t = i & 63;               // (coalesced 8-float runs per row)
    const float v =
        x[(((size_t)bimg * C_DIM + c) * H_IMG + h0 + (t >> 3)) * W_IMG +
          w0 + (t & 7)];
    sm.u.xT[t * C_DIM + c] = (__bf16)v;
  }
  // zero-pad d = 48..63 of qT / kT (K-dim padding for the Q^T K WMMA)
  for (int i = tid; i < NH * TT * 16; i += 256) {
    const int hh = i >> 10;
    const int t  = (i >> 4) & 63;
    const int j  = i & 15;
    sm.qT[hh][t * 64 + 48 + j] = (__bf16)0.0f;
    sm.kT[hh][t * 64 + 48 + j] = (__bf16)0.0f;
  }
  if (tid < QKV_ROWS) sm.bq[tid] = b_qkv[tid];   // 576 of 256? (looped below)
  for (int i = tid + 256; i < QKV_ROWS; i += 256) sm.bq[i] = b_qkv[i];
  if (tid < C_DIM) sm.bo[tid] = b_out[tid];
  __syncthreads();

  // ---- Phase 2: QKV = Wqkv @ x  (M=576, N=64, K=192) ----
  // Wave owns one M(o)-tile; 4 accumulators span all N(t)-tiles; A-frag
  // reused 4x; k-chunks double-buffered in registers for load/WMMA overlap.
  for (int mo = wave; mo < 36; mo += 8) {
    const int o0 = mo * 16;
    v8f acc[4] = {v8f{}, v8f{}, v8f{}, v8f{}};
    v16bf a_cur = load_a_frag(wq, o0, 0, C_DIM, lane);
    v16bf b_cur[4];
#pragma unroll
    for (int j = 0; j < 4; ++j)
      b_cur[j] = load_b_frag(sm.u.xT, j * 16, 0, C_DIM, lane);
#pragma unroll
    for (int kk = 0; kk < 6; ++kk) {
      v16bf a_nxt = a_cur;
      v16bf b_nxt[4] = {b_cur[0], b_cur[1], b_cur[2], b_cur[3]};
      if (kk < 5) {
        a_nxt = load_a_frag(wq, o0, (kk + 1) * 32, C_DIM, lane);
#pragma unroll
        for (int j = 0; j < 4; ++j)
          b_nxt[j] = load_b_frag(sm.u.xT, j * 16, (kk + 1) * 32, C_DIM, lane);
      }
#pragma unroll
      for (int j = 0; j < 4; ++j) acc[j] = WMMA_BF16(a_cur, b_cur[j], acc[j]);
      a_cur = a_nxt;
#pragma unroll
      for (int j = 0; j < 4; ++j) b_cur[j] = b_nxt[j];
    }
    // scatter D tiles into per-head q/k/v LDS layouts (+bias, q pre-scaled)
#pragma unroll
    for (int j = 0; j < 4; ++j) {
      const int t = j * 16 + fn;
#pragma unroll
      for (int r = 0; r < 8; ++r) {
        const int o  = o0 + r + 8 * fh;
        const float val = acc[j][r] + sm.bq[o];
        if (o < C_DIM) {                       // Q (pre-scaled)
          sm.qT[o / HD][t * 64 + (o % HD)] = (__bf16)(val * SCALE);
        } else if (o < 2 * C_DIM) {            // K
          const int oo = o - C_DIM;
          sm.kT[oo / HD][t * 64 + (oo % HD)] = (__bf16)val;
        } else {                               // V, stored [d][t]
          const int oo = o - 2 * C_DIM;
          sm.vv[oo / HD][(oo % HD) * TT + t] = (__bf16)val;
        }
      }
    }
  }
  __syncthreads();

  // ---- Phase 3: per-head attention ----
  for (int head = 0; head < NH; ++head) {
    // S = Q^T K : M=N=64, K=64(padded). Wave owns one M-tile + 2 N-tiles.
    {
      const int mq  = (wave >> 1) * 16;
      const int nk0 = (wave & 1) * 32;
      v8f acc[2] = {v8f{}, v8f{}};
#pragma unroll
      for (int kk = 0; kk < 2; ++kk) {
        v16bf a = load_a_frag(sm.qT[head], mq, kk * 32, 64, lane);
#pragma unroll
        for (int j = 0; j < 2; ++j) {
          v16bf bb = load_b_frag(sm.kT[head], nk0 + j * 16, kk * 32, 64, lane);
          acc[j] = WMMA_BF16(a, bb, acc[j]);
        }
      }
#pragma unroll
      for (int j = 0; j < 2; ++j)
#pragma unroll
        for (int r = 0; r < 8; ++r)
          sm.u.s[(mq + r + 8 * fh) * 65 + nk0 + j * 16 + fn] = acc[j][r];
    }
    __syncthreads();

    // softmax: 4 threads per row, partials combined via shfl_xor(1,2)
    {
      const int row = tid >> 2;
      const int qtr = tid & 3;
      const float* srow = &sm.u.s[row * 65 + qtr * 16];
      float e[16];
      float mx = srow[0];
#pragma unroll
      for (int j = 1; j < 16; ++j) mx = fmaxf(mx, srow[j]);
      mx = fmaxf(mx, __shfl_xor(mx, 1, 32));
      mx = fmaxf(mx, __shfl_xor(mx, 2, 32));
      float sum = 0.0f;
#pragma unroll
      for (int j = 0; j < 16; ++j) {
        e[j] = __expf(srow[j] - mx);
        sum += e[j];
      }
      sum += __shfl_xor(sum, 1, 32);
      sum += __shfl_xor(sum, 2, 32);
      const float inv = 1.0f / sum;
#pragma unroll
      for (int j = 0; j < 16; ++j)
        sm.p[row * TT + qtr * 16 + j] = (__bf16)(e[j] * inv);
    }
    __syncthreads();

    // O = P V^T : M=64, N=48, K=64. Wave owns one M-tile + 1-2 N-tiles.
    {
      const int mt   = (wave >> 1) * 16;
      const int half = wave & 1;
      const int nd0  = half ? 2 : 0;
      const int ndc  = half ? 1 : 2;
      v8f acc[2] = {v8f{}, v8f{}};
#pragma unroll
      for (int kk = 0; kk < 2; ++kk) {
        v16bf a = load_a_frag(sm.p, mt, kk * 32, 64, lane);
        for (int j = 0; j < ndc; ++j) {
          v16bf bb =
              load_b_frag(sm.vv[head], (nd0 + j) * 16, kk * 32, TT, lane);
          acc[j] = WMMA_BF16(a, bb, acc[j]);
        }
      }
      for (int j = 0; j < ndc; ++j)
#pragma unroll
        for (int r = 0; r < 8; ++r)
          sm.attn[(mt + r + 8 * fh) * C_DIM + head * HD + (nd0 + j) * 16 +
                  fn] = (__bf16)acc[j][r];
    }
    __syncthreads();   // before s/p reuse by next head
  }

  // ---- Phase 4: Y = attn @ Wout^T  (M=64, N=192, K=192) ----
  // Wave owns one M(t)-tile and 6 N(o)-tiles: A-frag reused 6x, k-chunks
  // double-buffered. wo (row-major [o][c]) is exactly B^T.
  {
    const int mt  = (wave >> 1) * 16;
    const int no0 = (wave & 1) * 96;
    v8f acc[6] = {v8f{}, v8f{}, v8f{}, v8f{}, v8f{}, v8f{}};
    v16bf a_cur = load_a_frag(sm.attn, mt, 0, C_DIM, lane);
    v16bf b_cur[6];
#pragma unroll
    for (int j = 0; j < 6; ++j)
      b_cur[j] = load_b_frag(wo, no0 + j * 16, 0, C_DIM, lane);
#pragma unroll
    for (int kk = 0; kk < 6; ++kk) {
      v16bf a_nxt = a_cur;
      v16bf b_nxt[6] = {b_cur[0], b_cur[1], b_cur[2],
                        b_cur[3], b_cur[4], b_cur[5]};
      if (kk < 5) {
        a_nxt = load_a_frag(sm.attn, mt, (kk + 1) * 32, C_DIM, lane);
#pragma unroll
        for (int j = 0; j < 6; ++j)
          b_nxt[j] = load_b_frag(wo, no0 + j * 16, (kk + 1) * 32, C_DIM, lane);
      }
#pragma unroll
      for (int j = 0; j < 6; ++j) acc[j] = WMMA_BF16(a_cur, b_cur[j], acc[j]);
      a_cur = a_nxt;
#pragma unroll
      for (int j = 0; j < 6; ++j) b_cur[j] = b_nxt[j];
    }
#pragma unroll
    for (int j = 0; j < 6; ++j) {
      const int o  = no0 + j * 16 + fn;
      const float bov = sm.bo[o];
#pragma unroll
      for (int r = 0; r < 8; ++r) {
        const int t = mt + r + 8 * fh;
        out[(((size_t)bimg * C_DIM + o) * H_IMG + h0 + (t >> 3)) * W_IMG +
            w0 + (t & 7)] = acc[j][r] + bov;
      }
    }
  }
}

// ---------------------------------------------------------------------------
extern "C" void kernel_launch(void* const* d_in, const int* in_sizes, int n_in,
                              void* d_out, int out_size, void* d_ws,
                              size_t ws_size, hipStream_t stream) {
  (void)in_sizes; (void)n_in; (void)out_size; (void)ws_size;
  const float* x      = (const float*)d_in[0];
  const float* w_qkv  = (const float*)d_in[1];
  const float* b_qkv  = (const float*)d_in[2];
  const float* w_out  = (const float*)d_in[3];
  const float* b_out  = (const float*)d_in[4];
  float* out = (float*)d_out;

  __bf16* wq_b = (__bf16*)d_ws;                       // 576*192 bf16
  __bf16* wo_b = wq_b + (size_t)QKV_ROWS * C_DIM;     // 192*192 bf16

  const int conv_elems = QKV_ROWS * C_DIM;            // dominant count
  convert_weights_kernel<<<(conv_elems + 255) / 256, 256, 0, stream>>>(
      w_qkv, w_out, wq_b, wo_b);

  const int n_windows = B_IMG * (H_IMG / WS) * (W_IMG / WS);  // 8192
  win_attn_kernel<<<n_windows, 256, sizeof(Smem), stream>>>(
      x, b_qkv, b_out, wq_b, wo_b, out);
}
